// DLI_loss_full_6614249636368
// MI455X (gfx1250) — compile-verified, matrix operand order checked by reading, not run on hardware
//
#include <hip/hip_runtime.h>
#include <hip/hip_bf16.h>
#include <math.h>

// Problem dims (fixed by the reference)
#define BDIM 128
#define TDIM 384
#define ENCD 256
#define HIDD 256
#define G4   1024   // 4*HID
#define KC   512    // ENC + HID  (fused input+recurrent K)

typedef __bf16 bf16v16 __attribute__((ext_vector_type(16)));
typedef float  f32v8   __attribute__((ext_vector_type(8)));

#if __has_builtin(__builtin_amdgcn_global_load_async_to_lds_b32) && \
    __has_builtin(__builtin_amdgcn_s_wait_asynccnt)
#define HAVE_ASYNC_LDS 1
typedef __attribute__((address_space(1))) int* g32_ptr;   // global src (per diag: __device__ int*)
typedef __attribute__((address_space(3))) int* l32_ptr;   // LDS dst
#endif

union FragU { bf16v16 v; uint4 q[2]; };

__device__ __forceinline__ unsigned short f2bf(float f){
  unsigned int u = __float_as_uint(f);
  u += 0x7FFFu + ((u >> 16) & 1u);       // round-to-nearest-even
  return (unsigned short)(u >> 16);
}

// One 16x32 bf16 fragment = 16 bf16 per lane = two contiguous 16B chunks
// (ISA 7.12.2: half-wave holds K {kb..kb+7} then {kb+16..kb+23})
__device__ __forceinline__ bf16v16 load_frag(const unsigned short* rowbase, int kb){
  FragU f;
  f.q[0] = *reinterpret_cast<const uint4*>(rowbase + kb);
  f.q[1] = *reinterpret_cast<const uint4*>(rowbase + kb + 16);
  return f.v;
}

__device__ __forceinline__ float sigm(float x){ return 1.0f / (1.0f + __expf(-x)); }

// ---------------------------------------------------------------------------
// Kernel 1: pack [W_ih | W_hh] -> bf16 Wcat[1024][512], bcomb = b_ih + b_hh
// ---------------------------------------------------------------------------
__global__ void prep_kernel(const float* __restrict__ Wih, const float* __restrict__ Whh,
                            const float* __restrict__ bih, const float* __restrict__ bhh,
                            unsigned short* __restrict__ Wcat, float* __restrict__ bcomb){
  int idx = blockIdx.x * 256 + threadIdx.x;           // 0 .. 1024*512-1
  if (idx < G4 * KC){
    int g = idx >> 9, k = idx & 511;
    float v = (k < ENCD) ? Wih[g * ENCD + k] : Whh[g * HIDD + (k - ENCD)];
    Wcat[idx] = f2bf(v);
  }
  if (idx < G4) bcomb[idx] = bih[idx] + bhh[idx];
}

// ---------------------------------------------------------------------------
// Kernel 2: xw[b,k] = encoder_output[b,k,:] . w_fc[HID:]   (one wave per (b,k))
// ---------------------------------------------------------------------------
__global__ void xw_kernel(const float* __restrict__ x, const float* __restrict__ wfc,
                          float* __restrict__ xw){
  int wid  = (blockIdx.x * blockDim.x + threadIdx.x) >> 5;
  int lane = threadIdx.x & 31;
  if (wid >= BDIM * TDIM) return;
  const float* row = x  + (size_t)wid * ENCD;
  const float* wx  = wfc + HIDD;
  float s = 0.0f;
  #pragma unroll
  for (int j = 0; j < 8; ++j){
    int e = lane + 32 * j;
    s += row[e] * wx[e];
  }
  #pragma unroll
  for (int m = 16; m >= 1; m >>= 1) s += __shfl_xor(s, m, 32);
  if (lane == 0) xw[wid] = s;
}

// ---------------------------------------------------------------------------
// Kernel 3 (optional, workspace-permitting): xg = x . W_ih^T  as a full-chip
// WMMA GEMM.  M = B*T = 49152 rows, N = 1024, K = ENC = 256.  One WG (16
// waves) per 16-row M tile; wave wv owns 4 N-tiles (64 columns).
// ---------------------------------------------------------------------------
__global__ __launch_bounds__(512) void xg_kernel(
    const float* __restrict__ x, const unsigned short* __restrict__ Wcat,
    float* __restrict__ xg)
{
  extern __shared__ unsigned short smx[];              // bf16 [16][ENCD]
  const int tid = threadIdx.x, lane = tid & 31, wv = tid >> 5;
  const size_t mbase = (size_t)blockIdx.x * 16;

  for (int e = tid; e < 16 * ENCD; e += 512){
    int r = e >> 8, k = e & 255;
    smx[r * ENCD + k] = f2bf(x[(mbase + r) * ENCD + k]);
  }
  __syncthreads();

  const int n    = lane & 15;
  const int khi  = (lane >> 4) * 8;
  const int mrow = (lane >> 4) * 8;
  const int colbase0 = wv * 64;
  const unsigned short* Arow = smx + (size_t)(lane & 15) * ENCD;
  const unsigned short* Br0 = Wcat + (size_t)(colbase0 +  0 + n) * KC;
  const unsigned short* Br1 = Wcat + (size_t)(colbase0 + 16 + n) * KC;
  const unsigned short* Br2 = Wcat + (size_t)(colbase0 + 32 + n) * KC;
  const unsigned short* Br3 = Wcat + (size_t)(colbase0 + 48 + n) * KC;

  f32v8 a0 = {}, a1 = {}, a2 = {}, a3 = {};
  #pragma unroll
  for (int k0 = 0; k0 < ENCD; k0 += 32){
    bf16v16 af = load_frag(Arow, k0 + khi);
    bf16v16 b0 = load_frag(Br0,  k0 + khi);
    bf16v16 b1 = load_frag(Br1,  k0 + khi);
    bf16v16 b2 = load_frag(Br2,  k0 + khi);
    bf16v16 b3 = load_frag(Br3,  k0 + khi);
    a0 = __builtin_amdgcn_wmma_f32_16x16x32_bf16(false, af, false, b0, (short)0, a0, false, false);
    a1 = __builtin_amdgcn_wmma_f32_16x16x32_bf16(false, af, false, b1, (short)0, a1, false, false);
    a2 = __builtin_amdgcn_wmma_f32_16x16x32_bf16(false, af, false, b2, (short)0, a2, false, false);
    a3 = __builtin_amdgcn_wmma_f32_16x16x32_bf16(false, af, false, b3, (short)0, a3, false, false);
  }
  #pragma unroll
  for (int j = 0; j < 8; ++j){
    size_t m = mbase + mrow + j;
    xg[m * G4 + colbase0 +  0 + n] = a0[j];
    xg[m * G4 + colbase0 + 16 + n] = a1[j];
    xg[m * G4 + colbase0 + 32 + n] = a2[j];
    xg[m * G4 + colbase0 + 48 + n] = a3[j];
  }
}

// ---------------------------------------------------------------------------
// Kernel 4: persistent WMMA LSTM.  8 WGs x 512 threads; WG b handles batch
// rows [16b,16b+16).  If use_xg, the input projection is added during the
// elementwise phase (coalesced) and the recurrence runs K=256; else fused
// K=512.  K start offset rotates with t so the L2-resident weight-fragment
// loads cannot be hoisted (and spilled).
// ---------------------------------------------------------------------------
__global__ __launch_bounds__(512) void lstm_kernel(
    const float* __restrict__ x, const unsigned short* __restrict__ Wcat,
    const float* __restrict__ bcomb, const float* __restrict__ wfc,
    const float* __restrict__ bfc, const float* __restrict__ xg,
    int use_xg, float* __restrict__ hw)
{
  extern __shared__ unsigned short smem[];
  unsigned short* As     = smem;                         // bf16 [16][KC]  (x_t | h)
  float*          gatesF = (float*)(smem + 16 * KC);     // f32  [16][G4]

  const int tid  = threadIdx.x;
  const int lane = tid & 31;
  const int wv   = tid >> 5;               // wave id 0..15
  const int bb   = blockIdx.x * 16;        // first batch row of this WG

  // zero recurrent half of A (h = 0 at t = 0)
  for (int e = tid; e < 16 * HIDD; e += 512){
    int r = e >> 8, k = e & 255;
    As[r * KC + ENCD + k] = 0;
  }

  // Elementwise role: wave wv owns batch row wv; lane owns h-indices lane+32j
  float c[8], bi[8], bff[8], bg[8], bo[8], wh[8];
  #pragma unroll
  for (int j = 0; j < 8; ++j){
    int hidx = lane + 32 * j;
    c[j]   = 0.0f;
    bi[j]  = bcomb[hidx];
    bff[j] = bcomb[HIDD     + hidx];
    bg[j]  = bcomb[2 * HIDD + hidx];
    bo[j]  = bcomb[3 * HIDD + hidx];
    wh[j]  = wfc[hidx];
  }
  const float bfc0 = bfc[0];

  // WMMA role: wave wv owns gate columns [wv*64, wv*64+64) = 4 N-tiles
  const int colbase0 = wv * 64;
  const int n    = lane & 15;
  const int khi  = (lane >> 4) * 8;        // half-wave K sub-offset (0 or 8)
  const int mrow = (lane >> 4) * 8;        // D-row offset for this half-wave
  const unsigned short* Arow  = As + (size_t)(lane & 15) * KC;
  const unsigned short* Brow0 = Wcat + (size_t)(colbase0 +  0 + n) * KC;
  const unsigned short* Brow1 = Wcat + (size_t)(colbase0 + 16 + n) * KC;
  const unsigned short* Brow2 = Wcat + (size_t)(colbase0 + 32 + n) * KC;
  const unsigned short* Brow3 = Wcat + (size_t)(colbase0 + 48 + n) * KC;

  __syncthreads();                         // h=0 init visible to all waves

  for (int t = 0; t < TDIM; ++t){
    // ---- Phase A (fused path only): stage x_t into As[:,0:ENC] as bf16 ----
    if (!use_xg){
      for (int e = tid; e < 16 * ENCD; e += 512){
        int r = e >> 8, k = e & 255;
        const float* xp = x + ((size_t)(bb + r) * TDIM + t) * ENCD + k;
        As[r * KC + k] = f2bf(*xp);
        if (t + 1 < TDIM) __builtin_prefetch(xp + ENCD, 0, 1);
      }
      __syncthreads();
    }

    // ---- Phase B: 16x1024 gate tile via WMMA (zero-init accumulate) ----
    f32v8 a0 = {}, a1 = {}, a2 = {}, a3 = {};
    if (use_xg){
      const int rot = (t * 32) & 255;                 // defeat LICM of B loads
      #pragma unroll
      for (int i = 0; i < 8; ++i){
        int k0 = ENCD + ((rot + i * 32) & 255);
        bf16v16 af = load_frag(Arow,  k0 + khi);
        bf16v16 b0 = load_frag(Brow0, k0 + khi);
        bf16v16 b1 = load_frag(Brow1, k0 + khi);
        bf16v16 b2 = load_frag(Brow2, k0 + khi);
        bf16v16 b3 = load_frag(Brow3, k0 + khi);
        a0 = __builtin_amdgcn_wmma_f32_16x16x32_bf16(false, af, false, b0, (short)0, a0, false, false);
        a1 = __builtin_amdgcn_wmma_f32_16x16x32_bf16(false, af, false, b1, (short)0, a1, false, false);
        a2 = __builtin_amdgcn_wmma_f32_16x16x32_bf16(false, af, false, b2, (short)0, a2, false, false);
        a3 = __builtin_amdgcn_wmma_f32_16x16x32_bf16(false, af, false, b3, (short)0, a3, false, false);
      }
    } else {
      const int rot = (t * 32) & 511;                 // defeat LICM of B loads
      #pragma unroll
      for (int i = 0; i < 16; ++i){
        int k0 = (rot + i * 32) & 511;
        bf16v16 af = load_frag(Arow,  k0 + khi);
        bf16v16 b0 = load_frag(Brow0, k0 + khi);
        bf16v16 b1 = load_frag(Brow1, k0 + khi);
        bf16v16 b2 = load_frag(Brow2, k0 + khi);
        bf16v16 b3 = load_frag(Brow3, k0 + khi);
        a0 = __builtin_amdgcn_wmma_f32_16x16x32_bf16(false, af, false, b0, (short)0, a0, false, false);
        a1 = __builtin_amdgcn_wmma_f32_16x16x32_bf16(false, af, false, b1, (short)0, a1, false, false);
        a2 = __builtin_amdgcn_wmma_f32_16x16x32_bf16(false, af, false, b2, (short)0, a2, false, false);
        a3 = __builtin_amdgcn_wmma_f32_16x16x32_bf16(false, af, false, b3, (short)0, a3, false, false);
      }
    }
    #pragma unroll
    for (int j = 0; j < 8; ++j){
      int m = mrow + j;                    // D layout: VGPR j -> M = j + 8*hi
      gatesF[m * G4 + colbase0 +  0 + n] = a0[j];
      gatesF[m * G4 + colbase0 + 16 + n] = a1[j];
      gatesF[m * G4 + colbase0 + 32 + n] = a2[j];
      gatesF[m * G4 + colbase0 + 48 + n] = a3[j];
    }
    __syncthreads();

    // ---- Phase C: cell update for batch row wv; write h (bf16) back to A ----
    float hwp = 0.0f;
    const size_t xrow = ((size_t)(bb + wv) * TDIM + t) * G4;
    #pragma unroll
    for (int j = 0; j < 8; ++j){
      int hidx = lane + 32 * j;
      float xgi = 0.0f, xgf = 0.0f, xgg = 0.0f, xgo = 0.0f;
      if (use_xg){                          // coalesced: lanes read adjacent cols
        xgi = xg[xrow +            hidx];
        xgf = xg[xrow + HIDD     + hidx];
        xgg = xg[xrow + 2 * HIDD + hidx];
        xgo = xg[xrow + 3 * HIDD + hidx];
      }
      float gi = gatesF[wv * G4 +            hidx] + bi[j]  + xgi;
      float gf = gatesF[wv * G4 + HIDD     + hidx] + bff[j] + xgf;
      float gg = gatesF[wv * G4 + 2 * HIDD + hidx] + bg[j]  + xgg;
      float go = gatesF[wv * G4 + 3 * HIDD + hidx] + bo[j]  + xgo;
      float cn = sigm(gf) * c[j] + sigm(gi) * tanhf(gg);
      c[j] = cn;
      float h = sigm(go) * tanhf(cn);
      As[wv * KC + ENCD + hidx] = f2bf(h);
      hwp += h * wh[j];
    }
    #pragma unroll
    for (int m = 16; m >= 1; m >>= 1) hwp += __shfl_xor(hwp, m, 32);
    if (lane == 0) hw[(size_t)(bb + wv) * TDIM + t] = hwp + bfc0;
    __syncthreads();
  }
}

// ---------------------------------------------------------------------------
// Kernel 5: per-batch masked LSE loss partials.
// logits[b,t,k] = hw[b,t] + xw[b,k]; valid k in (t, len); tgt at k = t+1.
// sxw staging uses the CDNA5 async global->LDS path when available.
// ---------------------------------------------------------------------------
__global__ void loss_kernel(const int* __restrict__ mask, const float* __restrict__ hw,
                            const float* __restrict__ xw, float* __restrict__ partials){
  __shared__ float sxw[TDIM];
  __shared__ int   ired[256];
  __shared__ float fred[256];
  __shared__ int   slen;
  const int b = blockIdx.x, tid = threadIdx.x;

#ifdef HAVE_ASYNC_LDS
  for (int k = tid; k < TDIM; k += 256){
    __builtin_amdgcn_global_load_async_to_lds_b32(
        (g32_ptr)(xw + b * TDIM + k), (l32_ptr)&sxw[k], 0, 0);
  }
  __builtin_amdgcn_s_wait_asynccnt(0);
#else
  for (int k = tid; k < TDIM; k += 256) sxw[k] = xw[b * TDIM + k];
#endif

  int lp = 0;
  for (int k = tid; k < TDIM; k += 256) lp += mask[b * TDIM + k];
  ired[tid] = lp;
  __syncthreads();
  for (int s = 128; s >= 1; s >>= 1){ if (tid < s) ired[tid] += ired[tid + s]; __syncthreads(); }
  if (tid == 0) slen = ired[0];
  __syncthreads();
  const int len = slen;

  float ls = 0.0f, lc = 0.0f;
  for (int t = tid; t < TDIM; t += 256){
    if (t + 1 < len){
      float hwbt = hw[b * TDIM + t];
      float mx = -3.402823e38f;
      for (int k = t + 1; k < len; ++k) mx = fmaxf(mx, sxw[k]);
      float s = 0.0f;
      for (int k = t + 1; k < len; ++k) s += expf(sxw[k] - mx);
      float lse = hwbt + mx + logf(s);
      float tgt = hwbt + sxw[t + 1];
      ls += lse - tgt;
      lc += 1.0f;
    }
  }
  fred[tid] = ls;
  __syncthreads();
  for (int s = 128; s >= 1; s >>= 1){ if (tid < s) fred[tid] += fred[tid + s]; __syncthreads(); }
  if (tid == 0) partials[2 * b] = fred[0];
  __syncthreads();
  fred[tid] = lc;
  __syncthreads();
  for (int s = 128; s >= 1; s >>= 1){ if (tid < s) fred[tid] += fred[tid + s]; __syncthreads(); }
  if (tid == 0) partials[2 * b + 1] = fred[0];
}

// ---------------------------------------------------------------------------
// Kernel 6: final reduce of 128 (sum,count) pairs -> scalar mean
// ---------------------------------------------------------------------------
__global__ void final_kernel(const float* __restrict__ partials, float* __restrict__ out){
  __shared__ float s1[128], s2[128];
  int tid = threadIdx.x;
  s1[tid] = partials[2 * tid];
  s2[tid] = partials[2 * tid + 1];
  __syncthreads();
  for (int s = 64; s >= 1; s >>= 1){
    if (tid < s){ s1[tid] += s1[tid + s]; s2[tid] += s2[tid + s]; }
    __syncthreads();
  }
  if (tid == 0) out[0] = s1[0] / s2[0];
}

extern "C" void kernel_launch(void* const* d_in, const int* in_sizes, int n_in,
                              void* d_out, int out_size, void* d_ws, size_t ws_size,
                              hipStream_t stream) {
  (void)in_sizes; (void)n_in; (void)out_size;
  const float* x    = (const float*)d_in[0];
  const int*   mask = (const int*)  d_in[1];
  const float* Wih  = (const float*)d_in[2];
  const float* Whh  = (const float*)d_in[3];
  const float* bih  = (const float*)d_in[4];
  const float* bhh  = (const float*)d_in[5];
  const float* wfc  = (const float*)d_in[6];
  const float* bfc  = (const float*)d_in[7];

  char* ws = (char*)d_ws;
  unsigned short* Wcat  = (unsigned short*)(ws);             // 1,048,576 B
  float*          bcomb = (float*)(ws + 1048576);            //     4,096 B
  float*          hw    = (float*)(ws + 1052672);            //   196,608 B
  float*          xw    = (float*)(ws + 1249280);            //   196,608 B
  float*          parts = (float*)(ws + 1445888);            //     1,024 B
  const size_t XG_OFF   = 1446912;
  const size_t XG_BYTES = (size_t)BDIM * TDIM * G4 * 4;      // 201,326,592 B
  const int use_xg = (ws_size >= XG_OFF + XG_BYTES) ? 1 : 0;
  float* xg = use_xg ? (float*)(ws + XG_OFF) : nullptr;

  // 1) pack bf16 weights + combined bias
  prep_kernel<<<(G4 * KC + 255) / 256, 256, 0, stream>>>(Wih, Whh, bih, bhh, Wcat, bcomb);
  // 2) xw projection (one wave per (b,k))
  xw_kernel<<<(BDIM * TDIM) / 8, 256, 0, stream>>>(x, wfc, xw);
  // 3) optional full-chip WMMA GEMM for the input projection
  if (use_xg)
    xg_kernel<<<(BDIM * TDIM) / 16, 512, 16 * ENCD * 2, stream>>>(x, Wcat, xg);
  // 4) persistent WMMA LSTM (8 WGs x 16 waves), dynamic LDS = A(bf16) + gates(f32)
  lstm_kernel<<<8, 512, 16 * KC * 2 + 16 * G4 * 4, stream>>>(x, Wcat, bcomb, wfc, bfc,
                                                             xg, use_xg, hw);
  // 5) loss partials per batch row
  loss_kernel<<<BDIM, 256, 0, stream>>>(mask, hw, xw, parts);
  // 6) final scalar
  final_kernel<<<1, 128, 0, stream>>>(parts, (float*)d_out);
}
